// Mamba3MultiModal_71356586655998
// MI455X (gfx1250) — matrix operand order repaired and checked by reading, not compiled, not run on hardware
//
#include <hip/hip_runtime.h>
#include <hip/hip_bf16.h>

typedef unsigned short u16;
typedef __attribute__((ext_vector_type(4)))  unsigned int u32x4;
typedef __attribute__((ext_vector_type(16))) __bf16      v16bf;
typedef __attribute__((ext_vector_type(8)))  float       v8f;

#define D_    768
#define S_    1024
#define B_    8
#define L_    4
#define M_    (B_*S_)   /* 8192 tokens */
#define KCONV 4
#define NCHUNK 24       /* scan: 24 chunks of 32 channels */

__device__ __forceinline__ u16 f2bf(float f) {
  unsigned int u = __builtin_bit_cast(unsigned int, f);
  u += 0x7FFFu + ((u >> 16) & 1u);          // round-to-nearest-even
  return (u16)(u >> 16);
}

// ---------------- conversion / packing kernels ----------------
__global__ void k_f32_to_bf16(const float* __restrict__ src, u16* __restrict__ dst, int n) {
  int i = blockIdx.x * blockDim.x + threadIdx.x;
  if (i < n) dst[i] = f2bf(src[i]);
}

// pack Wd|WB|WC into [L][3D][D] bf16 so delta/B/C come from ONE gemm
__global__ void k_pack_dbc_w(const float* __restrict__ wd, const float* __restrict__ wb,
                             const float* __restrict__ wc, u16* __restrict__ dst) {
  int i = blockIdx.x * blockDim.x + threadIdx.x;
  if (i >= L_ * 3 * D_ * D_) return;
  int c = i % D_;
  int r = (i / D_) % (3 * D_);
  int l = i / (D_ * 3 * D_);
  float v;
  if      (r <   D_) v = wd[(size_t)(l*D_ + r       )*D_ + c];
  else if (r < 2*D_) v = wb[(size_t)(l*D_ + r -   D_)*D_ + c];
  else               v = wc[(size_t)(l*D_ + r - 2*D_)*D_ + c];
  dst[i] = f2bf(v);
}

__global__ void k_pack_dbc_b(const float* __restrict__ bd, const float* __restrict__ bb,
                             const float* __restrict__ bc, float* __restrict__ dst) {
  int i = blockIdx.x * blockDim.x + threadIdx.x;
  if (i >= L_ * 3 * D_) return;
  int r = i % (3 * D_), l = i / (3 * D_);
  float v;
  if (r < D_) v = bd[l*D_ + r];
  else if (r < 2*D_) v = bb[l*D_ + r - D_];
  else v = bc[l*D_ + r - 2*D_];
  dst[i] = v;
}

// ---------------- embedding ----------------
__global__ void k_embed(const int* __restrict__ tok, const float* __restrict__ emb,
                        const float* __restrict__ pos, float* __restrict__ x) {
  int i = blockIdx.x * blockDim.x + threadIdx.x;
  if (i >= M_ * D_) return;
  int d = i % D_, t = i / D_, s = t % S_;
  x[i] = emb[(size_t)tok[t] * D_ + d] + pos[(size_t)s * D_ + d];
}

// ---------------- block reduction helper (LayerNorm) ----------------
__device__ __forceinline__ float block_sum(float v, float* sm) {
  for (int o = 16; o > 0; o >>= 1) v += __shfl_down(v, o, 32);
  int lane = threadIdx.x & 31, w = threadIdx.x >> 5;
  if (lane == 0) sm[w] = v;
  __syncthreads();
  float r = 0.f;
  int nw = blockDim.x >> 5;
  if ((int)threadIdx.x < nw) r = sm[threadIdx.x];
  if (w == 0) {
    for (int o = 16; o > 0; o >>= 1) r += __shfl_down(r, o, 32);
    if (lane == 0) sm[0] = r;
  }
  __syncthreads();
  r = sm[0];
  __syncthreads();
  return r;
}

// ---------------- LayerNorm -> bf16 activations ----------------
__global__ __launch_bounds__(256)
void k_layernorm_bf16(const float* __restrict__ x, const float* __restrict__ g,
                      const float* __restrict__ b, u16* __restrict__ h) {
  __shared__ float sm[32];
  int t = blockIdx.x;
  const float* xr = x + (size_t)t * D_;
  float lv[3]; float s = 0.f;
  #pragma unroll
  for (int k = 0; k < 3; ++k) { lv[k] = xr[threadIdx.x + k*256]; s += lv[k]; }
  float mean = block_sum(s, sm) * (1.f / D_);
  float vs = 0.f;
  #pragma unroll
  for (int k = 0; k < 3; ++k) { float d = lv[k] - mean; vs += d * d; }
  float inv = rsqrtf(block_sum(vs, sm) * (1.f / D_) + 1e-5f);
  #pragma unroll
  for (int k = 0; k < 3; ++k) {
    int c = threadIdx.x + k*256;
    h[(size_t)t * D_ + c] = f2bf((lv[k] - mean) * inv * g[c] + b[c]);
  }
}

// ---------------- depthwise causal conv (K=4) ----------------
__global__ void k_dwconv(const float* __restrict__ u, const float* __restrict__ cw,
                         const float* __restrict__ cb, float* __restrict__ uc,
                         u16* __restrict__ ucbf) {
  int i = blockIdx.x * blockDim.x + threadIdx.x;
  if (i >= M_ * D_) return;
  int d = i % D_, t = i / D_, s = t % S_;
  float acc = cb[d];
  #pragma unroll
  for (int j = 0; j < KCONV; ++j) {
    int ss = s - (KCONV - 1) + j;
    if (ss >= 0) acc += u[(size_t)(t - (KCONV-1) + j) * D_ + d] * cw[d * KCONV + j];
  }
  uc[i] = acc;
  ucbf[i] = f2bf(acc);
}

// ---------------- selective scan: per-wave chunks, no block barriers ----------------
__global__ __launch_bounds__(256)
void k_scan(const float* __restrict__ decay, const float* __restrict__ inp,
            const float* __restrict__ cm, float* __restrict__ ypart) {
  int b = blockIdx.x / 3;
  int wave = threadIdx.x >> 5, lane = threadIdx.x & 31;
  int chunk = (blockIdx.x % 3) * 8 + wave;        // 0..23
  int n = chunk * 32 + lane;                      // channel
  size_t base = (size_t)b * S_ * D_ + n;
  float* yp = ypart + ((size_t)b * NCHUNK + chunk) * S_;
  float h = 0.f;
  for (int s = 0; s < S_; ++s) {
    size_t o = base + (size_t)s * D_;
    h = h * decay[o] + inp[o];
    float p = h * cm[o];
    #pragma unroll
    for (int off = 16; off > 0; off >>= 1) p += __shfl_down(p, off, 32);
    if (lane == 0) yp[s] = p;
  }
}

__global__ void k_yreduce(const float* __restrict__ ypart, float* __restrict__ y) {
  int i = blockIdx.x * blockDim.x + threadIdx.x;
  if (i >= M_) return;
  int b = i / S_, s = i % S_;
  float v = 0.f;
  #pragma unroll
  for (int c = 0; c < NCHUNK; ++c) v += ypart[((size_t)b * NCHUNK + c) * S_ + s];
  y[i] = v;
}

__global__ void k_gate(const float* __restrict__ y, const float* __restrict__ sigv,
                       u16* __restrict__ yg) {
  int i = blockIdx.x * blockDim.x + threadIdx.x;
  if (i >= M_ * D_) return;
  yg[i] = f2bf(y[i / D_] * sigv[i]);
}

// ---------------- bf16 WMMA GEMM: C[M,N] = A[M,K] * W[N,K]^T + bias ----------------
// 256 threads (8 waves), block tile 128x128, BK=32, double-buffered LDS.
// Wave tile 64(M) x 32(N) = 4x2 WMMA 16x16x32 per K-step.
#define MODE_UV  0
#define MODE_DBC 1
#define MODE_OUT 2

struct Frag { u32x4 q0, q1; };

template<int MODE>
__global__ __launch_bounds__(256)
void k_wmma_gemm(const u16* __restrict__ A, const u16* __restrict__ W,
                 const float* __restrict__ bias, int K,
                 float* __restrict__ p0, float* __restrict__ p1,
                 float* __restrict__ p2, const float* __restrict__ p3) {
  __shared__ __align__(16) u16 As[2][128 * 40];   // stride 40 halves = 80B (16B aligned)
  __shared__ __align__(16) u16 Ws[2][128 * 40];
  const int n0 = blockIdx.x * 128;
  const int m0 = blockIdx.y * 128;
  const int tid  = threadIdx.x;
  const int wave = tid >> 5, lane = tid & 31;
  const int wm = (wave >> 2) * 64;                // 0 or 64
  const int wn = (wave & 3) * 32;                 // 0,32,64,96
  const int fm = lane & 15, hi = lane >> 4;

  v8f acc[4][2] = {};
  const int NK = K >> 5;
  u32x4 ra[2], rw[2];

  auto gload = [&](int k0) {
    #pragma unroll
    for (int rep = 0; rep < 2; ++rep) {
      int idx = tid + rep * 256;                  // 0..511 chunks (16B each)
      int row = idx >> 2, seg = idx & 3;
      ra[rep] = *(const u32x4*)&A[(size_t)(m0 + row) * K + k0 + seg * 8];
      rw[rep] = *(const u32x4*)&W[(size_t)(n0 + row) * K + k0 + seg * 8];
    }
  };
  auto sstore = [&](int buf) {
    #pragma unroll
    for (int rep = 0; rep < 2; ++rep) {
      int idx = tid + rep * 256;
      int row = idx >> 2, seg = idx & 3;
      *(u32x4*)&As[buf][row * 40 + seg * 8] = ra[rep];
      *(u32x4*)&Ws[buf][row * 40 + seg * 8] = rw[rep];
    }
  };

  gload(0);
  sstore(0);
  __syncthreads();

  for (int k = 0; k < NK; ++k) {
    const int b = k & 1;
    if (k + 1 < NK) gload((k + 1) << 5);

    Frag fa[4], fb[2];
    #pragma unroll
    for (int i = 0; i < 4; ++i) {                 // A: lane fm = m; k-halves split by hi
      const u16* ap = &As[b][(wm + i*16 + fm) * 40];
      fa[i].q0 = *(const u32x4*)&ap[hi * 8];
      fa[i].q1 = *(const u32x4*)&ap[16 + hi * 8];
    }
    #pragma unroll
    for (int j = 0; j < 2; ++j) {                 // B: lane fm = n; k-halves split by hi
      const u16* wp = &Ws[b][(wn + j*16 + fm) * 40];
      fb[j].q0 = *(const u32x4*)&wp[hi * 8];
      fb[j].q1 = *(const u32x4*)&wp[16 + hi * 8];
    }
    if (k + 1 < NK) sstore(b ^ 1);                // opposite buffer: no WAR within this step

    #pragma unroll
    for (int i = 0; i < 4; ++i)
      #pragma unroll
      for (int j = 0; j < 2; ++j)
        acc[i][j] = __builtin_amdgcn_wmma_f32_16x16x32_bf16(
            false, __builtin_bit_cast(v16bf, fa[i]),
            false, __builtin_bit_cast(v16bf, fb[j]),
            (short)0, acc[i][j], false, false);

    __syncthreads();                              // single barrier per K-step
  }

  // epilogue: C/D layout -> lane fm = n, (hi*8 + r) = m
  #pragma unroll
  for (int i = 0; i < 4; ++i) {
    #pragma unroll
    for (int j = 0; j < 2; ++j) {
      int gn = n0 + wn + j*16 + fm;
      float bv = bias[gn];
      #pragma unroll
      for (int r = 0; r < 8; ++r) {
        int gm = m0 + wm + i*16 + hi*8 + r;
        float v = acc[i][j][r] + bv;
        if (MODE == MODE_UV) {
          if (gn < D_) p0[(size_t)gm * D_ + gn] = v;                             // u
          else         p1[(size_t)gm * D_ + gn - D_] = 1.f / (1.f + __expf(-v)); // sigmoid(v)
        } else if (MODE == MODE_DBC) {
          if (gn < D_)        p0[(size_t)gm * D_ + gn] = 1.f / (1.f + __expf(v)); // exp(-softplus)
          else if (gn < 2*D_) p1[(size_t)gm * D_ + gn - D_] = p3[(size_t)gm * D_ + gn - D_] * v; // uc*B
          else                p2[(size_t)gm * D_ + gn - 2*D_] = v;                // C
        } else {
          p0[(size_t)gm * D_ + gn] += v;                                          // residual add
        }
      }
    }
  }
}

// ---------------- host ----------------
static inline size_t alignup(size_t v) { return (v + 255) & ~(size_t)255; }

extern "C" void kernel_launch(void* const* d_in, const int* in_sizes, int n_in,
                              void* d_out, int out_size, void* d_ws, size_t ws_size,
                              hipStream_t stream) {
  const int*   tokens = (const int*)d_in[0];
  const float* emb    = (const float*)d_in[1];
  const float* pos    = (const float*)d_in[2];
  const float* ln_g   = (const float*)d_in[3];
  const float* ln_b   = (const float*)d_in[4];
  const float* Win    = (const float*)d_in[5];
  const float* b_in   = (const float*)d_in[6];
  const float* convw  = (const float*)d_in[7];
  const float* convb  = (const float*)d_in[8];
  const float* Wd     = (const float*)d_in[9];
  const float* bd     = (const float*)d_in[10];
  const float* WB     = (const float*)d_in[11];
  const float* bB     = (const float*)d_in[12];
  const float* WC     = (const float*)d_in[13];
  const float* bC     = (const float*)d_in[14];
  const float* Wout   = (const float*)d_in[15];
  const float* bout   = (const float*)d_in[16];
  float* x = (float*)d_out;

  char* ws = (char*)d_ws;
  size_t off = 0;
  auto alloc = [&](size_t bytes) -> char* { char* p = ws + off; off = alignup(off + bytes); return p; };

  u16*   win_bf  = (u16*)  alloc((size_t)L_ * 2*D_ * D_ * 2);
  u16*   wdbc_bf = (u16*)  alloc((size_t)L_ * 3*D_ * D_ * 2);
  u16*   wout_bf = (u16*)  alloc((size_t)L_ * D_ * D_ * 2);
  float* bdbc    = (float*)alloc((size_t)L_ * 3*D_ * 4);
  u16*   h_bf    = (u16*)  alloc((size_t)M_ * D_ * 2);
  float* u       = (float*)alloc((size_t)M_ * D_ * 4);
  float* sigv    = (float*)alloc((size_t)M_ * D_ * 4);
  float* uc      = (float*)alloc((size_t)M_ * D_ * 4);
  u16*   uc_bf   = (u16*)  alloc((size_t)M_ * D_ * 2);
  float* decay   = (float*)alloc((size_t)M_ * D_ * 4);
  float* inp     = (float*)alloc((size_t)M_ * D_ * 4);
  float* cm      = (float*)alloc((size_t)M_ * D_ * 4);
  float* ypart   = (float*)alloc((size_t)B_ * NCHUNK * S_ * 4);
  float* y       = (float*)alloc((size_t)M_ * 4);
  u16*   yg_bf   = (u16*)  alloc((size_t)M_ * D_ * 2);
  (void)ws_size; (void)in_sizes; (void)n_in; (void)out_size;

  const int TPB = 256;
  int nWin = L_ * 2*D_ * D_;
  k_f32_to_bf16<<<(nWin + TPB - 1)/TPB, TPB, 0, stream>>>(Win, win_bf, nWin);
  int nWout = L_ * D_ * D_;
  k_f32_to_bf16<<<(nWout + TPB - 1)/TPB, TPB, 0, stream>>>(Wout, wout_bf, nWout);
  int nDbc = L_ * 3*D_ * D_;
  k_pack_dbc_w<<<(nDbc + TPB - 1)/TPB, TPB, 0, stream>>>(Wd, WB, WC, wdbc_bf);
  k_pack_dbc_b<<<(L_*3*D_ + TPB - 1)/TPB, TPB, 0, stream>>>(bd, bB, bC, bdbc);

  int nMD = M_ * D_;
  k_embed<<<(nMD + TPB - 1)/TPB, TPB, 0, stream>>>(tokens, emb, pos, x);

  for (int l = 0; l < L_; ++l) {
    k_layernorm_bf16<<<M_, 256, 0, stream>>>(x, ln_g + l*D_, ln_b + l*D_, h_bf);

    dim3 g1(2*D_/128, M_/128);
    k_wmma_gemm<MODE_UV><<<g1, 256, 0, stream>>>(h_bf, win_bf + (size_t)l*2*D_*D_,
        b_in + l*2*D_, D_, u, sigv, nullptr, nullptr);

    k_dwconv<<<(nMD + TPB - 1)/TPB, TPB, 0, stream>>>(u, convw + l*D_*KCONV, convb + l*D_, uc, uc_bf);

    dim3 g2(3*D_/128, M_/128);
    k_wmma_gemm<MODE_DBC><<<g2, 256, 0, stream>>>(uc_bf, wdbc_bf + (size_t)l*3*D_*D_,
        bdbc + l*3*D_, D_, decay, inp, cm, uc);

    k_scan<<<B_*3, 256, 0, stream>>>(decay, inp, cm, ypart);
    k_yreduce<<<(M_ + TPB - 1)/TPB, TPB, 0, stream>>>(ypart, y);
    k_gate<<<(nMD + TPB - 1)/TPB, TPB, 0, stream>>>(y, sigv, yg_bf);

    dim3 g3(D_/128, M_/128);
    k_wmma_gemm<MODE_OUT><<<g3, 256, 0, stream>>>(yg_bf, wout_bf + (size_t)l*D_*D_,
        bout + l*D_, D_, x, nullptr, nullptr, nullptr);
  }
}
